// MultiHeadAttention_34093450395988
// MI455X (gfx1250) — matrix operand order repaired and checked by reading, not compiled, not run on hardware
//
#include <hip/hip_runtime.h>

// Problem constants (fixed by the reference)
#define BB   4
#define SS   2048
#define DD   1024
#define NH   16
#define HD   64
#define NTOK (BB * SS)   // 8192

typedef __attribute__((ext_vector_type(16))) __bf16 v16bf;
typedef __attribute__((ext_vector_type(8)))  float  v8f;
typedef __attribute__((ext_vector_type(4)))  unsigned int v4u;
typedef __attribute__((ext_vector_type(8)))  int v8i;
typedef __attribute__((ext_vector_type(4)))  int v4i;

static __device__ __forceinline__ v8f zero8() {
  v8f r = {0.f, 0.f, 0.f, 0.f, 0.f, 0.f, 0.f, 0.f};
  return r;
}

// ---------------- TDM: 2D bf16 tile -> LDS (row-padded) ----------------
// Loads tile_y rows x tile_x bf16 elements (contiguous), global row stride
// `stride_elems`, into LDS at lds_byte_addr with +8 bf16 padding per row
// (pad_interval = 32 DWORDs [code 4], pad_amount = 4 DWORDs [code 3]).
// Requires tile_x == 64 (row = 32 DWORDs) so padding lands at row ends.
static __device__ __forceinline__ void tdm_load_tile64(unsigned int lds_byte_addr,
                                                       const __bf16* gptr,
                                                       unsigned int tile_y,
                                                       unsigned long long stride_elems) {
  const unsigned long long ga = (unsigned long long)(uintptr_t)gptr;
  const unsigned int tile_x = 64u;
  v4u g0;
  g0[0] = 1u;                                           // count=1, user desc
  g0[1] = lds_byte_addr;                                // LDS byte address
  g0[2] = (unsigned int)(ga & 0xFFFFFFFFu);             // global_addr[31:0]
  g0[3] = (unsigned int)((ga >> 32) & 0x01FFFFFFu) | (2u << 30);  // [56:32] | type=2

  const unsigned int dw0 = (1u << 16)                   // data_size = 1 (2 bytes)
                         | (1u << 20)                   // pad_enable
                         | (4u << 22)                   // pad_interval: 32 DWORDs
                         | (3u << 25);                  // pad_amount: 4 DWORDs (8 bf16)
  v8i g1;
  g1[0] = (int)dw0;
  g1[1] = (int)((tile_x & 0xFFFFu) << 16);                            // tensor_dim0[15:0]
  g1[2] = (int)(((tile_x >> 16) & 0xFFFFu) | ((tile_y & 0xFFFFu) << 16)); // td0 hi | td1 lo
  g1[3] = (int)(((tile_y >> 16) & 0xFFFFu) | ((tile_x & 0xFFFFu) << 16)); // td1 hi | tile_dim0
  g1[4] = (int)(tile_y & 0xFFFFu);                                    // tile_dim1 (tile_dim2=0)
  g1[5] = (int)(stride_elems & 0xFFFFFFFFull);                        // dim0_stride[31:0]
  g1[6] = (int)((stride_elems >> 32) & 0xFFFFull);                    // dim0_stride[47:32]
  g1[7] = 0;
  v4i z4 = {0, 0, 0, 0};
#if __clang_major__ >= 23
  v8i z8 = {0, 0, 0, 0, 0, 0, 0, 0};
  __builtin_amdgcn_tensor_load_to_lds(g0, g1, z4, z4, z8, 0);
#else
  __builtin_amdgcn_tensor_load_to_lds(g0, g1, z4, z4, 0);
#endif
}

// A fragment, 16x32 bf16 (M x K), row-major source, leading dim `ld` elements.
static __device__ __forceinline__ v16bf load_frag_a(const __bf16* p0, int ld, int lane) {
  const int m = lane & 15, hh = lane >> 4;
  const __bf16* p = p0 + (size_t)m * ld + hh * 8;
  union { v16bf v; uint4 q[2]; } u;
  u.q[0] = *(const uint4*)(p);
  u.q[1] = *(const uint4*)(p + 16);
  return u.v;
}

// B fragment, 32x16 bf16 (K x N), from an N-major tile: row n holds K contiguously.
static __device__ __forceinline__ v16bf load_frag_b(const __bf16* p0, int ld, int lane) {
  const int n = lane & 15, hh = lane >> 4;
  const __bf16* p = p0 + (size_t)n * ld + hh * 8;
  union { v16bf v; uint4 q[2]; } u;
  u.q[0] = *(const uint4*)(p);
  u.q[1] = *(const uint4*)(p + 16);
  return u.v;
}

static __device__ __forceinline__ v8f wmma_bf16(v16bf a, v16bf b, v8f c) {
  return __builtin_amdgcn_wmma_f32_16x16x32_bf16(false, a, false, b, (short)0, c, false, false);
}

// ---------------- elementwise prep kernels ----------------

__global__ void mha_cast_bf16(const float* __restrict__ s, __bf16* __restrict__ d, int n) {
  int i = blockIdx.x * 256 + threadIdx.x;
  if (i < n) d[i] = (__bf16)s[i];
}

// d[n*DD + k] = (bf16) s[k*DD + n]
__global__ void mha_transpose_cast(const float* __restrict__ s, __bf16* __restrict__ d) {
  int i = blockIdx.x * 256 + threadIdx.x;
  int n = i >> 10, k = i & (DD - 1);
  d[i] = (__bf16)s[(size_t)k * DD + n];
}

// ------- 128x128-tile bf16 GEMM, TDM double-buffered:  C = A[M,1024] @ W  (Wt = W^T) -------

template <bool F32OUT>
__global__ __launch_bounds__(256) void mha_gemm128(const __bf16* __restrict__ A,
                                                   const __bf16* __restrict__ Wt,
                                                   void* __restrict__ out,
                                                   const float* __restrict__ bias) {
  __shared__ __bf16 As[2][128][72];  // 128 rows x 64 k (+8 pad), double-buffered
  __shared__ __bf16 Bs[2][128][72];  // 128 cols(n) x 64 k (+8 pad)
  const int tid  = threadIdx.x;
  const int lane = tid & 31;
  const int wav  = tid >> 5;           // 8 waves: 2 (m) x 4 (n)
  const int wm   = (wav >> 2) * 64;    // 64-row strip
  const int wn   = (wav & 3) * 32;     // 32-col strip (2 x 16)
  const int m0   = blockIdx.y * 128;
  const int n0   = blockIdx.x * 128;

  v8f acc[4][2];
#pragma unroll
  for (int i = 0; i < 4; ++i) { acc[i][0] = zero8(); acc[i][1] = zero8(); }

  const unsigned int ldsA0 = (unsigned int)(uintptr_t)&As[0][0][0];
  const unsigned int ldsA1 = (unsigned int)(uintptr_t)&As[1][0][0];
  const unsigned int ldsB0 = (unsigned int)(uintptr_t)&Bs[0][0][0];
  const unsigned int ldsB1 = (unsigned int)(uintptr_t)&Bs[1][0][0];

  // prologue: DMA k-slab 0 into buffer 0
  if (wav == 0) {
    tdm_load_tile64(ldsA0, A  + (size_t)m0 * DD, 128u, DD);
    tdm_load_tile64(ldsB0, Wt + (size_t)n0 * DD, 128u, DD);
    __builtin_amdgcn_s_wait_tensorcnt(0);
  }
  __syncthreads();

  for (int kt = 0; kt < 16; ++kt) {        // 16 k-slabs of 64
    const int cur = kt & 1;
    if (wav == 0 && kt + 1 < 16) {         // overlap next slab DMA with compute
      const int k1 = (kt + 1) * 64;
      tdm_load_tile64(cur ? ldsA0 : ldsA1, A  + (size_t)m0 * DD + k1, 128u, DD);
      tdm_load_tile64(cur ? ldsB0 : ldsB1, Wt + (size_t)n0 * DD + k1, 128u, DD);
    }
#pragma unroll
    for (int ks = 0; ks < 2; ++ks) {       // two 32-wide WMMA k-steps
      v16bf bf0 = load_frag_b(&Bs[cur][wn][ks * 32],      72, lane);
      v16bf bf1 = load_frag_b(&Bs[cur][wn + 16][ks * 32], 72, lane);
#pragma unroll
      for (int i = 0; i < 4; ++i) {
        v16bf af = load_frag_a(&As[cur][wm + i * 16][ks * 32], 72, lane);
        acc[i][0] = wmma_bf16(af, bf0, acc[i][0]);
        acc[i][1] = wmma_bf16(af, bf1, acc[i][1]);
      }
    }
    if (wav == 0 && kt + 1 < 16) __builtin_amdgcn_s_wait_tensorcnt(0);
    __syncthreads();
  }

  const int nl = lane & 15, hh = lane >> 4;
#pragma unroll
  for (int j = 0; j < 2; ++j) {
    const int col = n0 + wn + j * 16 + nl;
#pragma unroll
    for (int i = 0; i < 4; ++i) {
#pragma unroll
      for (int r = 0; r < 8; ++r) {
        const size_t row = (size_t)(m0 + wm + i * 16 + r + 8 * hh);
        if constexpr (F32OUT) {
          ((float*)out)[row * DD + col] = acc[i][j][r] + bias[col];
        } else {
          ((__bf16*)out)[row * DD + col] = (__bf16)acc[i][j][r];
        }
      }
    }
  }
}

// ---------------- causal flash attention (per b,h), bf16 WMMA, f32 softmax ----------------

__global__ __launch_bounds__(128) void mha_flash_attn(const __bf16* __restrict__ Qb,
                                                      const __bf16* __restrict__ Kb,
                                                      const __bf16* __restrict__ Vb,
                                                      __bf16* __restrict__ ctx) {
  __shared__ __bf16 Kt[64][72];  // [key][dim]     (TDM-staged, row-padded)
  __shared__ __bf16 Vt[64][72];  // [dim][key]     (manual transpose for P@V B-fragments)
  __shared__ __bf16 Pt[64][72];  // [q_local][key_local]

  const int tid  = threadIdx.x;
  const int lane = tid & 31;
  const int wq   = tid >> 5;               // wave -> 16-query strip
  const int bh   = blockIdx.y;
  const int bb   = bh / NH, h = bh % NH;
  const int q0   = blockIdx.x * 64;
  const int qg0  = q0 + wq * 16;
  const int nl   = lane & 15, hh = lane >> 4;

  const unsigned int ldsK = (unsigned int)(uintptr_t)&Kt[0][0];

  // Q fragments for this wave (A-layout), K = 64 dims -> two k-steps of 32.
  const __bf16* qp = Qb + ((size_t)(bb * SS + qg0)) * DD + h * HD;
  const v16bf qf0 = load_frag_a(qp,      DD, lane);
  const v16bf qf1 = load_frag_a(qp + 32, DD, lane);

  float m_i[8], l_i[8];
#pragma unroll
  for (int r = 0; r < 8; ++r) { m_i[r] = -3e38f; l_i[r] = 0.f; }
  v8f o[4];
#pragma unroll
  for (int t = 0; t < 4; ++t) o[t] = zero8();

  const int key2 = tid >> 1;     // 0..63 : key row this thread stages (V)
  const int half = tid & 1;      // 32-dim half

  for (int j0 = 0; j0 <= q0 + 63; j0 += 64) {
    __syncthreads();  // previous iteration's consumers are done with Kt/Vt
    if (wq == 0) {    // DMA the K tile (64 keys x 64 dims, stride D) into padded LDS
      tdm_load_tile64(ldsK, Kb + ((size_t)(bb * SS + j0)) * DD + h * HD, 64u, DD);
    }
    {  // manual transposed staging of V
      const size_t vrow = ((size_t)(bb * SS + j0 + key2)) * DD + h * HD + half * 32;
      union { uint4 q[4]; __bf16 e[32]; } vbuf;
      const uint4* vs = (const uint4*)(Vb + vrow);
      vbuf.q[0] = vs[0]; vbuf.q[1] = vs[1]; vbuf.q[2] = vs[2]; vbuf.q[3] = vs[3];
#pragma unroll
      for (int i = 0; i < 32; ++i) Vt[half * 32 + i][key2] = vbuf.e[i];
    }
    if (wq == 0) __builtin_amdgcn_s_wait_tensorcnt(0);
    __syncthreads();

    // S = Q @ K^T  (4 key subtiles x 2 k-steps)
    v8f sc[4];
#pragma unroll
    for (int t = 0; t < 4; ++t) sc[t] = zero8();
#pragma unroll
    for (int t = 0; t < 4; ++t) {
      v16bf b0 = load_frag_b(&Kt[t * 16][0],  72, lane);
      v16bf b1 = load_frag_b(&Kt[t * 16][32], 72, lane);
      sc[t] = wmma_bf16(qf0, b0, sc[t]);
      sc[t] = wmma_bf16(qf1, b1, sc[t]);
    }

    // scale (1/sqrt(64)=0.125), causal mask, online softmax
    float mnew[8], alpha[8];
#pragma unroll
    for (int r = 0; r < 8; ++r) {
      const int q = qg0 + r + 8 * hh;
      float rm = -3e38f;
#pragma unroll
      for (int t = 0; t < 4; ++t) {
        float s = sc[t][r] * 0.125f;
        if (j0 + t * 16 + nl > q) s = -1e30f;
        sc[t][r] = s;
        rm = fmaxf(rm, s);
      }
      rm = fmaxf(rm, __shfl_xor(rm, 1, 16));
      rm = fmaxf(rm, __shfl_xor(rm, 2, 16));
      rm = fmaxf(rm, __shfl_xor(rm, 4, 16));
      rm = fmaxf(rm, __shfl_xor(rm, 8, 16));
      mnew[r]  = fmaxf(m_i[r], rm);
      alpha[r] = __expf(m_i[r] - mnew[r]);
      m_i[r]   = mnew[r];
    }

#pragma unroll
    for (int r = 0; r < 8; ++r) {
      float rs = 0.f;
#pragma unroll
      for (int t = 0; t < 4; ++t) {
        float p = __expf(sc[t][r] - mnew[r]);
        rs += p;
        Pt[wq * 16 + r + 8 * hh][t * 16 + nl] = (__bf16)p;
      }
      rs += __shfl_xor(rs, 1, 16);
      rs += __shfl_xor(rs, 2, 16);
      rs += __shfl_xor(rs, 4, 16);
      rs += __shfl_xor(rs, 8, 16);
      l_i[r] = l_i[r] * alpha[r] + rs;
#pragma unroll
      for (int t = 0; t < 4; ++t) o[t][r] *= alpha[r];
    }

    // same-wave LDS store->load ordering for Pt
    asm volatile("s_wait_dscnt 0x0" ::: "memory");

    // O += P @ V   (A = Pt rows of this wave, B = Vt[dim][key])
    v16bf pa0 = load_frag_a(&Pt[wq * 16][0],  72, lane);
    v16bf pa1 = load_frag_a(&Pt[wq * 16][32], 72, lane);
#pragma unroll
    for (int t = 0; t < 4; ++t) {
      v16bf vb0 = load_frag_b(&Vt[t * 16][0],  72, lane);
      v16bf vb1 = load_frag_b(&Vt[t * 16][32], 72, lane);
      o[t] = wmma_bf16(pa0, vb0, o[t]);
      o[t] = wmma_bf16(pa1, vb1, o[t]);
    }
  }

  // normalize and write ctx (bf16, token x D)
#pragma unroll
  for (int t = 0; t < 4; ++t) {
#pragma unroll
    for (int r = 0; r < 8; ++r) {
      const size_t tok = (size_t)(bb * SS + qg0 + r + 8 * hh);
      const int col = h * HD + t * 16 + nl;
      ctx[tok * DD + col] = (__bf16)(o[t][r] / l_i[r]);
    }
  }
}

// ---------------- host orchestration ----------------

extern "C" void kernel_launch(void* const* d_in, const int* in_sizes, int n_in,
                              void* d_out, int out_size, void* d_ws, size_t ws_size,
                              hipStream_t stream) {
  (void)in_sizes; (void)n_in; (void)out_size; (void)ws_size;
  const float* x  = (const float*)d_in[0];
  const float* Wq = (const float*)d_in[1];
  const float* Wk = (const float*)d_in[2];
  const float* Wv = (const float*)d_in[3];
  const float* Wo = (const float*)d_in[4];
  const float* bo = (const float*)d_in[5];
  float* out = (float*)d_out;

  char* ws = (char*)d_ws;
  const size_t xbytes = (size_t)NTOK * DD * 2;   // 16 MB
  const size_t wbytes = (size_t)DD * DD * 2;     // 2 MB
  __bf16* xb   = (__bf16*)ws;                ws += xbytes;
  __bf16* wqT  = (__bf16*)ws;                ws += wbytes;
  __bf16* wkT  = (__bf16*)ws;                ws += wbytes;
  __bf16* wvT  = (__bf16*)ws;                ws += wbytes;
  __bf16* woT  = (__bf16*)ws;                ws += wbytes;
  __bf16* qb   = (__bf16*)ws;                ws += xbytes;
  __bf16* kb   = (__bf16*)ws;                ws += xbytes;
  __bf16* vb   = (__bf16*)ws;                ws += xbytes;
  __bf16* ctxb = (__bf16*)ws;                ws += xbytes;

  // 1) precision prep
  mha_cast_bf16<<<(NTOK * DD) / 256, 256, 0, stream>>>(x, xb, NTOK * DD);
  mha_transpose_cast<<<(DD * DD) / 256, 256, 0, stream>>>(Wq, wqT);
  mha_transpose_cast<<<(DD * DD) / 256, 256, 0, stream>>>(Wk, wkT);
  mha_transpose_cast<<<(DD * DD) / 256, 256, 0, stream>>>(Wv, wvT);
  mha_transpose_cast<<<(DD * DD) / 256, 256, 0, stream>>>(Wo, woT);

  // 2) QKV projections (TDM double-buffered 128x128 tiles)
  dim3 gg(DD / 128, NTOK / 128);
  mha_gemm128<false><<<gg, 256, 0, stream>>>(xb, wqT, qb, nullptr);
  mha_gemm128<false><<<gg, 256, 0, stream>>>(xb, wkT, kb, nullptr);
  mha_gemm128<false><<<gg, 256, 0, stream>>>(xb, wvT, vb, nullptr);

  // 3) causal flash attention
  dim3 ga(SS / 64, BB * NH);
  mha_flash_attn<<<ga, 128, 0, stream>>>(qb, kb, vb, ctxb);

  // 4) output projection + bias (f32 out)
  mha_gemm128<true><<<gg, 256, 0, stream>>>(ctxb, woT, (void*)out, bo);
}
